// HHNOneAttention_77773267796105
// MI455X (gfx1250) — matrix operand order repaired and checked by reading, not compiled, not run on hardware
//
#include <hip/hip_runtime.h>

// ---------------- problem constants ----------------
#define NN    10000
#define EE    100000
#define NHEADS 4
#define GG    8
#define SPG   1250
#define SPGP  1280
#define INW   16
#define PEW   5
#define EDW   8
#define OUTW  4
#define NL    5
#define QT    79          // ceil(1250/16)

#define CEILD(a,b) (((a)+(b)-1)/(b))

// ---------------- types ----------------
typedef __attribute__((ext_vector_type(16))) __bf16       v16bf;
typedef __attribute__((ext_vector_type(8)))  float        v8f;
typedef __attribute__((ext_vector_type(4)))  unsigned int u32x4;

union FragAB { v16bf v; u32x4 q[2]; unsigned int u[8]; };
union FragC  { v8f   v; float f[8]; };

__device__ __forceinline__ __bf16 f2bf(float f) {
  unsigned u = __builtin_bit_cast(unsigned, f);
  unsigned r = u + 0x7FFFu + ((u >> 16) & 1u);   // round-to-nearest-even
  unsigned short s = (unsigned short)(r >> 16);
  return __builtin_bit_cast(__bf16, s);
}

// =====================================================================
// Templated WMMA bf16 GEMM:  out = act(A @ W + bias) [+ residual]
// A = up to NSEG row-gatherable segments of compile-time width SEGW
// (fuses the [h_src|h_dst|h_edge] / [h|msg] concats into fragment loads).
// W pre-transposed bf16 [Ntot, K] -> contiguous 32B B-fragment loads.
// Block = 8 waves; each wave computes 32M x 64N (2 A-frags, 8 f32
// accumulators, B fragments loaded once and reused by both M-subtiles:
// 12 b128 loads per 8 WMMAs). K-loop fully unrolled at compile time so
// all loads are global_load_b128 with immediate offsets (no flat path).
// =====================================================================
template <int SEGW, int NSEG, bool FULLN>
__global__ __launch_bounds__(256) void wmma_gemm_t(
    const __bf16* __restrict__ A0, const int* __restrict__ idx0,
    const __bf16* __restrict__ A1, const int* __restrict__ idx1,
    const __bf16* __restrict__ A2, const int* __restrict__ idx2,
    const __bf16* __restrict__ Wt,
    const float* __restrict__ bias,
    const float* __restrict__ residual,
    float* __restrict__ outF,
    __bf16* __restrict__ outB,
    const int* __restrict__ rowMask, int maskVal,
    int M, int Ntot, int relu)
{
  constexpr int K = SEGW * NSEG;
  const int lane = threadIdx.x & 31;
  const int wave = threadIdx.x >> 5;
  const int m0   = blockIdx.x * 256 + wave * 32;
  if (m0 >= M) return;                       // wave-uniform: M is a multiple of 16
  const int n0   = blockIdx.y * 64;
  const int half = lane >> 4;                // 0 or 1
  const int cl   = lane & 15;
  const bool mt1 = (m0 + 16) < M;            // second M-subtile valid (wave-uniform)
  const int r0   = m0 + cl;                  // A row for subtile 0
  int r1 = m0 + 16 + cl; if (r1 > M - 1) r1 = M - 1;   // clamped row for subtile 1

  // A-layout per lane: lane<16 -> K{0..7,16..23}, lane>=16 -> K{8..15,24..31}
  const __bf16* rp[2][3];
  rp[0][0] = A0 + (size_t)(idx0 ? idx0[r0] : r0) * SEGW + half * 8;
  rp[1][0] = A0 + (size_t)(idx0 ? idx0[r1] : r1) * SEGW + half * 8;
  if (NSEG > 1) {
    rp[0][1] = A1 + (size_t)(idx1 ? idx1[r0] : r0) * SEGW + half * 8;
    rp[1][1] = A1 + (size_t)(idx1 ? idx1[r1] : r1) * SEGW + half * 8;
  }
  if (NSEG > 2) {
    rp[0][2] = A2 + (size_t)(idx2 ? idx2[r0] : r0) * SEGW + half * 8;
    rp[1][2] = A2 + (size_t)(idx2 ? idx2[r1] : r1) * SEGW + half * 8;
  }

  // B-layout per lane: col = n0+j*16+cl, K {half*8..+7, 16+half*8..+7} (matches A)
  const __bf16* bp[4];
  bool bval[4];
#pragma unroll
  for (int j = 0; j < 4; ++j) {
    const int c = n0 + j * 16 + cl;
    bval[j] = FULLN || (c < Ntot);
    bp[j] = Wt + (size_t)(bval[j] ? c : 0) * K + half * 8;
  }

  FragC acc[2][4];
#pragma unroll
  for (int i = 0; i < 2; ++i)
#pragma unroll
    for (int j = 0; j < 4; ++j)
#pragma unroll
      for (int v = 0; v < 8; ++v) acc[i][j].f[v] = 0.f;

#pragma unroll
  for (int s = 0; s < NSEG; ++s) {
#pragma unroll
    for (int kl = 0; kl < SEGW; kl += 32) {
      const int k0 = s * SEGW + kl;
      FragAB a[2];
#pragma unroll
      for (int i = 0; i < 2; ++i) {
        a[i].q[0] = *(const u32x4*)(rp[i][s] + kl);
        a[i].q[1] = *(const u32x4*)(rp[i][s] + kl + 16);
      }
#pragma unroll
      for (int j = 0; j < 4; ++j) {
        FragAB b;
        b.q[0] = *(const u32x4*)(bp[j] + k0);
        b.q[1] = *(const u32x4*)(bp[j] + k0 + 16);
        if (!FULLN && !bval[j]) {
#pragma unroll
          for (int u = 0; u < 8; ++u) b.u[u] = 0u;
        }
        acc[0][j].v = __builtin_amdgcn_wmma_f32_16x16x32_bf16(
            false, a[0].v, false, b.v, (short)0, acc[0][j].v, false, false);
        acc[1][j].v = __builtin_amdgcn_wmma_f32_16x16x32_bf16(
            false, a[1].v, false, b.v, (short)0, acc[1][j].v, false, false);
      }
    }
  }

  // epilogue: D layout row = v + half*8 (+16 for subtile 1), col = lane&15
#pragma unroll
  for (int i = 0; i < 2; ++i) {
    if (i == 1 && !mt1) continue;            // wave-uniform skip
#pragma unroll
    for (int j = 0; j < 4; ++j) {
      const int c = n0 + j * 16 + cl;
      if (!FULLN && c >= Ntot) continue;
      const float bi = bias ? bias[c] : 0.f;
      const int rb = m0 + i * 16 + half * 8;
#pragma unroll
      for (int v = 0; v < 8; ++v) {
        float val = acc[i][j].f[v] + bi;
        if (relu) val = fmaxf(val, 0.f);
        const size_t o = (size_t)(rb + v) * Ntot + c;
        if (residual) val += residual[o];
        if (rowMask && rowMask[rb + v] != maskVal) continue;
        if (outF) outF[o] = val;
        if (outB) outB[o] = f2bf(val);
      }
    }
  }
}

// =====================================================================
// Flash attention over equal-size graphs: one wave per (g, head, q-tile).
// S = Q K^T via WMMA; online softmax with 16-lane shfl reductions;
// P re-laid-out D->A through per-wave LDS (s_wait_dscnt); O += P V via
// WMMA against pre-transposed zero-padded Vt.
// =====================================================================
__global__ __launch_bounds__(256) void attn_kernel(
    const __bf16* __restrict__ qkv,   // [N, 768] bf16
    const __bf16* __restrict__ Vt,    // [G*4*64, 1280] bf16 (zero padded)
    __bf16* __restrict__ o)           // [N, 256] bf16
{
  __shared__ __align__(16) __bf16 pshared[8][2][16][16];
  const int lane = threadIdx.x & 31;
  const int wave = threadIdx.x >> 5;
  const int task = blockIdx.x * 8 + wave;
  if (task >= GG * NHEADS * QT) return;
  const int qt = task % QT;
  const int gh = task / QT;
  const int h  = gh & 3;
  const int g  = gh >> 2;
  const int half = lane >> 4;
  const int cl   = lane & 15;
  const int q0   = qt * 16;

  int qi = q0 + cl; if (qi > SPG - 1) qi = SPG - 1;   // clamp padding rows
  const __bf16* qrow = qkv + (size_t)(g * SPG + qi) * 768 + h * 64;
  FragAB qa[2];
#pragma unroll
  for (int t = 0; t < 2; ++t) {                       // A layout: {0..7,16..23} split
    const __bf16* p = qrow + t * 32 + half * 8;
    qa[t].q[0] = *(const u32x4*)p;
    qa[t].q[1] = *(const u32x4*)(p + 16);
  }
  const __bf16* vbase = Vt + (size_t)((g * NHEADS + h) * 64) * SPGP;

  float mrow[8], lrow[8];
  FragC oacc[4];
#pragma unroll
  for (int v = 0; v < 8; ++v) { mrow[v] = -1e30f; lrow[v] = 0.f; }
#pragma unroll
  for (int j = 0; j < 4; ++j)
#pragma unroll
    for (int v = 0; v < 8; ++v) oacc[j].f[v] = 0.f;

  for (int kv0 = 0; kv0 < SPG; kv0 += 32) {
    // prefetch next V tile chunk for this lane's column stream (near-cache scope)
    __builtin_prefetch((const void*)(vbase + (size_t)cl * SPGP + kv0 + 32), 0, 3);
    FragC S[2];
#pragma unroll
    for (int t = 0; t < 2; ++t) {
      const int kvc = kv0 + t * 16 + cl;
      const int ki  = (kvc > SPG - 1) ? (SPG - 1) : kvc;
      const __bf16* krow = qkv + (size_t)(g * SPG + ki) * 768 + 256 + h * 64;
#pragma unroll
      for (int v = 0; v < 8; ++v) S[t].f[v] = 0.f;
#pragma unroll
      for (int ds = 0; ds < 2; ++ds) {                // B layout: {0..7,16..23} split
        FragAB kb;
        const __bf16* p = krow + ds * 32 + half * 8;
        kb.q[0] = *(const u32x4*)p;
        kb.q[1] = *(const u32x4*)(p + 16);
        S[t].v = __builtin_amdgcn_wmma_f32_16x16x32_bf16(
            false, qa[ds].v, false, kb.v, (short)0, S[t].v, false, false);
      }
      const bool valid = kvc < SPG;
#pragma unroll
      for (int v = 0; v < 8; ++v)
        S[t].f[v] = valid ? (S[t].f[v] * 0.125f) : -1e30f;   // 1/sqrt(64)
    }
    // online softmax: row = v + half*8, spread over 16 lanes of same half
    float mnew[8], alpha[8];
#pragma unroll
    for (int v = 0; v < 8; ++v) {
      float tm = fmaxf(S[0].f[v], S[1].f[v]);
      for (int off = 1; off < 16; off <<= 1) tm = fmaxf(tm, __shfl_xor(tm, off, 32));
      mnew[v]  = fmaxf(mrow[v], tm);
      alpha[v] = __expf(mrow[v] - mnew[v]);
      mrow[v]  = mnew[v];
    }
#pragma unroll
    for (int t = 0; t < 2; ++t)
#pragma unroll
      for (int v = 0; v < 8; ++v)
        S[t].f[v] = __expf(S[t].f[v] - mnew[v]);
#pragma unroll
    for (int v = 0; v < 8; ++v) {
      float s = S[0].f[v] + S[1].f[v];
      for (int off = 1; off < 16; off <<= 1) s += __shfl_xor(s, off, 32);
      lrow[v] = lrow[v] * alpha[v] + s;
#pragma unroll
      for (int j = 0; j < 4; ++j) oacc[j].f[v] *= alpha[v];
    }
    // P: D layout -> LDS -> A layout
#pragma unroll
    for (int t = 0; t < 2; ++t)
#pragma unroll
      for (int v = 0; v < 8; ++v)
        pshared[wave][t][v + half * 8][cl] = f2bf(S[t].f[v]);
    __asm__ volatile("s_wait_dscnt 0x0" ::: "memory");
    FragAB pa;   // A row = cl; K halves 0..7 from tile0, 8..15 from tile1
    {
      const __bf16* pb0 = &pshared[wave][0][cl][half * 8];
      const __bf16* pb1 = &pshared[wave][1][cl][half * 8];
      pa.q[0] = *(const u32x4*)pb0;
      pa.q[1] = *(const u32x4*)pb1;
    }
#pragma unroll
    for (int j = 0; j < 4; ++j) {
      FragAB vb;
      const __bf16* p = vbase + (size_t)(j * 16 + cl) * SPGP + kv0 + half * 8;
      vb.q[0] = *(const u32x4*)p;
      vb.q[1] = *(const u32x4*)(p + 16);
      oacc[j].v = __builtin_amdgcn_wmma_f32_16x16x32_bf16(
          false, pa.v, false, vb.v, (short)0, oacc[j].v, false, false);
    }
  }

#pragma unroll
  for (int j = 0; j < 4; ++j)
#pragma unroll
    for (int v = 0; v < 8; ++v) {
      const int q = q0 + v + half * 8;
      if (q < SPG) {
        const float val = oacc[j].f[v] / lrow[v];
        o[(size_t)(g * SPG + q) * 256 + h * 64 + j * 16 + cl] = f2bf(val);
      }
    }
}

// ---------------- support kernels ----------------
__global__ void wt_prep(const float* __restrict__ W, __bf16* __restrict__ Wt,
                        int Kin, int Kpad, int Nw) {
  const int i = blockIdx.x * 256 + threadIdx.x;
  if (i >= Nw * Kpad) return;
  const int n = i / Kpad, k = i - n * Kpad;
  const float v = (k < Kin) ? W[(size_t)k * Nw + n] : 0.f;
  Wt[i] = f2bf(v);
}

__global__ void ntype_kernel(const float* __restrict__ x, int* __restrict__ t) {
  const int i = blockIdx.x * 256 + threadIdx.x;
  if (i >= NN) return;
  float v = rintf(x[(size_t)i * INW + 2] * 3.0f);
  v = fminf(fmaxf(v, 1.f), 3.f);
  t[i] = (int)v - 1;
}

__global__ void pad2_kernel(const float* __restrict__ a, int aw,
                            const float* __restrict__ b, int bw,
                            __bf16* __restrict__ d, int dw, long rows) {
  const long i = (long)blockIdx.x * 256 + threadIdx.x;
  if (i >= rows * dw) return;
  const long r = i / dw;
  const int  k = (int)(i - r * dw);
  float v = 0.f;
  if (k < aw) v = a[r * aw + k];
  else if (b && k < aw + bw) v = b[r * bw + (k - aw)];
  d[i] = f2bf(v);
}

__global__ void zero_kernel(float* __restrict__ p, long n) {
  const long i = (long)blockIdx.x * 256 + threadIdx.x;
  if (i < n) p[i] = 0.f;
}

__global__ void cast_kernel(const float* __restrict__ s, __bf16* __restrict__ d, long n) {
  const long i = (long)blockIdx.x * 256 + threadIdx.x;
  if (i < n) d[i] = f2bf(s[i]);
}

__global__ void addcast_kernel(const float* __restrict__ a, const float* __restrict__ b,
                               __bf16* __restrict__ d, long n) {
  const long i = (long)blockIdx.x * 256 + threadIdx.x;
  if (i < n) d[i] = f2bf(a[i] + b[i]);
}

__global__ void segsum_kernel(const float* __restrict__ he, const int* __restrict__ dst,
                              float* __restrict__ m) {
  const long t = (long)blockIdx.x * 256 + threadIdx.x;
  const long e = t >> 6;
  const int  c = (int)(t & 63) * 4;
  if (e >= EE) return;
  const int d = dst[e];
  const float* p = he + (size_t)e * 256 + c;
  float* q = m + (size_t)d * 256 + c;
  atomicAdd(q + 0, p[0]); atomicAdd(q + 1, p[1]);
  atomicAdd(q + 2, p[2]); atomicAdd(q + 3, p[3]);
}

__global__ void vt_kernel(const __bf16* __restrict__ qkv, __bf16* __restrict__ vt) {
  const int i = blockIdx.x * 256 + threadIdx.x;
  const int tot = GG * NHEADS * 64 * SPGP;
  if (i >= tot) return;
  const int s   = i % SPGP;
  int rest      = i / SPGP;
  const int dch = rest % 64;   rest /= 64;
  const int h   = rest % NHEADS;
  const int g   = rest / NHEADS;
  __bf16 v = f2bf(0.f);
  if (s < SPG) v = qkv[(size_t)(g * SPG + s) * 768 + 512 + h * 64 + dch];
  vt[i] = v;
}

// =====================================================================
// host-side orchestration
// =====================================================================
extern "C" void kernel_launch(void* const* d_in, const int* in_sizes, int n_in,
                              void* d_out, int out_size, void* d_ws, size_t ws_size,
                              hipStream_t stream) {
  const float* x      = (const float*)d_in[0];
  const float* pe     = (const float*)d_in[1];
  const float* ea     = (const float*)d_in[2];
  const int*   ei     = (const int*)d_in[3];
  const float* enc_w1 = (const float*)d_in[5];
  const float* enc_b1 = (const float*)d_in[6];
  const float* enc_w2 = (const float*)d_in[7];
  const float* enc_b2 = (const float*)d_in[8];
  const float* ee_w1  = (const float*)d_in[9];
  const float* ee_b1  = (const float*)d_in[10];
  const float* ee_w2  = (const float*)d_in[11];
  const float* ee_b2  = (const float*)d_in[12];
  const float* eu_w1  = (const float*)d_in[13];
  const float* eu_b1  = (const float*)d_in[14];
  const float* eu_w2  = (const float*)d_in[15];
  const float* eu_b2  = (const float*)d_in[16];
  const float* nu_w1  = (const float*)d_in[17];
  const float* nu_b1  = (const float*)d_in[18];
  const float* nu_w2  = (const float*)d_in[19];
  const float* nu_b2  = (const float*)d_in[20];
  const float* qkv_w  = (const float*)d_in[21];
  const float* qkv_b  = (const float*)d_in[22];
  const float* out_w  = (const float*)d_in[23];
  const float* out_b  = (const float*)d_in[24];
  const float* fu_w1  = (const float*)d_in[25];
  const float* fu_b1  = (const float*)d_in[26];
  const float* fu_w2  = (const float*)d_in[27];
  const float* fu_b2  = (const float*)d_in[28];
  const float* dec_w1 = (const float*)d_in[29];
  const float* dec_b1 = (const float*)d_in[30];
  const float* dec_w2 = (const float*)d_in[31];
  const float* dec_b2 = (const float*)d_in[32];
  const int* src = ei;
  const int* dst = ei + EE;
  float* out = (float*)d_out;

  char* base = (char*)d_ws;
  size_t off = 0;
  auto alloc = [&](size_t bytes) -> void* {
    void* p = base + off;
    off = (off + bytes + 255) & ~(size_t)255;
    return p;
  };
  auto prep = [&](const float* W, int Kin, int Kpad, int Nw) -> __bf16* {
    __bf16* dp = (__bf16*)alloc((size_t)Nw * Kpad * sizeof(__bf16));
    const int tot = Nw * Kpad;
    wt_prep<<<CEILD(tot, 256), 256, 0, stream>>>(W, dp, Kin, Kpad, Nw);
    return dp;
  };

  // --- transposed bf16 weights ---
  __bf16 *enc_w1t[3], *enc_w2t[3], *dec_w1t[3], *dec_w2t[3];
  for (int t = 0; t < 3; ++t) enc_w1t[t] = prep(enc_w1 + (size_t)t * 21 * 256, 21, 32, 256);
  for (int t = 0; t < 3; ++t) enc_w2t[t] = prep(enc_w2 + (size_t)t * 256 * 256, 256, 256, 256);
  __bf16* ee_w1t = prep(ee_w1, 8, 32, 256);
  __bf16* ee_w2t = prep(ee_w2, 256, 256, 256);
  __bf16 *eu_w1t[NL], *eu_w2t[NL], *nu_w1t[NL], *nu_w2t[NL];
  for (int l = 0; l < NL; ++l) {
    eu_w1t[l] = prep(eu_w1 + (size_t)l * 768 * 256, 768, 768, 256);
    eu_w2t[l] = prep(eu_w2 + (size_t)l * 256 * 256, 256, 256, 256);
    nu_w1t[l] = prep(nu_w1 + (size_t)l * 512 * 256, 512, 512, 256);
    nu_w2t[l] = prep(nu_w2 + (size_t)l * 256 * 256, 256, 256, 256);
  }
  __bf16* qkv_wt = prep(qkv_w, 256, 256, 768);
  __bf16* out_wt = prep(out_w, 256, 256, 256);
  __bf16* fu_w1t = prep(fu_w1, 256, 256, 256);
  __bf16* fu_w2t = prep(fu_w2, 256, 256, 256);
  for (int t = 0; t < 3; ++t) dec_w1t[t] = prep(dec_w1 + (size_t)t * 256 * 256, 256, 256, 256);
  for (int t = 0; t < 3; ++t) dec_w2t[t] = prep(dec_w2 + (size_t)t * 256 * 4, 256, 256, 4);

  // --- activation buffers ---
  int*    ntype = (int*)alloc((size_t)NN * 4);
  __bf16* ninp  = (__bf16*)alloc((size_t)NN * 32 * 2);
  __bf16* einp  = (__bf16*)alloc((size_t)EE * 32 * 2);
  __bf16* tmpN  = (__bf16*)alloc((size_t)NN * 256 * 2);
  __bf16* tmpE  = (__bf16*)alloc((size_t)EE * 256 * 2);
  float*  hn_f  = (float*) alloc((size_t)NN * 256 * 4);
  __bf16* hn_b  = (__bf16*)alloc((size_t)NN * 256 * 2);
  float*  he_f  = (float*) alloc((size_t)EE * 256 * 4);
  __bf16* he_b  = (__bf16*)alloc((size_t)EE * 256 * 2);
  float*  mn_f  = (float*) alloc((size_t)NN * 256 * 4);
  __bf16* mn_b  = (__bf16*)alloc((size_t)NN * 256 * 2);
  __bf16* qkvb  = (__bf16*)alloc((size_t)NN * 768 * 2);
  __bf16* vtb   = (__bf16*)alloc((size_t)GG * NHEADS * 64 * SPGP * 2);
  __bf16* aob   = (__bf16*)alloc((size_t)NN * 256 * 2);
  float*  glob  = (float*) alloc((size_t)NN * 256 * 4);
  __bf16* fusei = (__bf16*)alloc((size_t)NN * 256 * 2);
  __bf16* hfin  = (__bf16*)alloc((size_t)NN * 256 * 2);

  // template dispatch over (segW, nseg, fullN)
  auto gemm = [&](const __bf16* A0, const int* i0, const __bf16* A1, const int* i1,
                  const __bf16* A2, const int* i2, int segW, int nseg,
                  const __bf16* Wt, const float* bias, const float* resid,
                  float* oF, __bf16* oB, const int* mask, int mval,
                  int M, int Ntot, int relu) {
    dim3 g(CEILD(M, 256), CEILD(Ntot, 64));
    if (segW == 32) {
      wmma_gemm_t<32, 1, true><<<g, 256, 0, stream>>>(A0, i0, A1, i1, A2, i2, Wt, bias,
          resid, oF, oB, mask, mval, M, Ntot, relu);
    } else if (nseg == 3) {
      wmma_gemm_t<256, 3, true><<<g, 256, 0, stream>>>(A0, i0, A1, i1, A2, i2, Wt, bias,
          resid, oF, oB, mask, mval, M, Ntot, relu);
    } else if (nseg == 2) {
      wmma_gemm_t<256, 2, true><<<g, 256, 0, stream>>>(A0, i0, A1, i1, A2, i2, Wt, bias,
          resid, oF, oB, mask, mval, M, Ntot, relu);
    } else if ((Ntot & 63) == 0) {
      wmma_gemm_t<256, 1, true><<<g, 256, 0, stream>>>(A0, i0, A1, i1, A2, i2, Wt, bias,
          resid, oF, oB, mask, mval, M, Ntot, relu);
    } else {
      wmma_gemm_t<256, 1, false><<<g, 256, 0, stream>>>(A0, i0, A1, i1, A2, i2, Wt, bias,
          resid, oF, oB, mask, mval, M, Ntot, relu);
    }
  };

  // --- input prep ---
  ntype_kernel<<<CEILD(NN, 256), 256, 0, stream>>>(x, ntype);
  pad2_kernel<<<CEILD((long)NN * 32, 256), 256, 0, stream>>>(x, INW, pe, PEW, ninp, 32, NN);
  pad2_kernel<<<CEILD((long)EE * 32, 256), 256, 0, stream>>>(ea, EDW, nullptr, 0, einp, 32, EE);

  // --- node encoder: 3 type-MLPs, row-masked store selects per node ---
  for (int t = 0; t < 3; ++t) {
    gemm(ninp, nullptr, nullptr, nullptr, nullptr, nullptr, 32, 1, enc_w1t[t],
         enc_b1 + t * 256, nullptr, nullptr, tmpN, nullptr, 0, NN, 256, 1);
    gemm(tmpN, nullptr, nullptr, nullptr, nullptr, nullptr, 256, 1, enc_w2t[t],
         enc_b2 + t * 256, nullptr, hn_f, hn_b, ntype, t, NN, 256, 0);
  }
  // --- edge encoder ---
  gemm(einp, nullptr, nullptr, nullptr, nullptr, nullptr, 32, 1, ee_w1t,
       ee_b1, nullptr, nullptr, tmpE, nullptr, 0, EE, 256, 1);
  gemm(tmpE, nullptr, nullptr, nullptr, nullptr, nullptr, 256, 1, ee_w2t,
       ee_b2, nullptr, he_f, he_b, nullptr, 0, EE, 256, 0);

  // --- message passing layers ---
  for (int l = 0; l < NL; ++l) {
    // edge update: fused gather [h_src | h_dst | h_edge] K=768
    gemm(hn_b, src, hn_b, dst, he_b, nullptr, 256, 3, eu_w1t[l],
         eu_b1 + l * 256, nullptr, nullptr, tmpE, nullptr, 0, EE, 256, 1);
    gemm(tmpE, nullptr, nullptr, nullptr, nullptr, nullptr, 256, 1, eu_w2t[l],
         eu_b2 + l * 256, he_f, he_f, he_b, nullptr, 0, EE, 256, 0);
    // aggregate messages
    zero_kernel<<<CEILD((long)NN * 256, 256), 256, 0, stream>>>(mn_f, (long)NN * 256);
    segsum_kernel<<<CEILD((long)EE * 64, 256), 256, 0, stream>>>(he_f, dst, mn_f);
    cast_kernel<<<CEILD((long)NN * 256, 256), 256, 0, stream>>>(mn_f, mn_b, (long)NN * 256);
    // node update: fused gather [h | m] K=512
    gemm(hn_b, nullptr, mn_b, nullptr, nullptr, nullptr, 256, 2, nu_w1t[l],
         nu_b1 + l * 256, nullptr, nullptr, tmpN, nullptr, 0, NN, 256, 1);
    gemm(tmpN, nullptr, nullptr, nullptr, nullptr, nullptr, 256, 1, nu_w2t[l],
         nu_b2 + l * 256, hn_f, hn_f, hn_b, nullptr, 0, NN, 256, 0);
  }

  // --- global attention ---
  gemm(hn_b, nullptr, nullptr, nullptr, nullptr, nullptr, 256, 1, qkv_wt,
       qkv_b, nullptr, nullptr, qkvb, nullptr, 0, NN, 768, 0);
  vt_kernel<<<CEILD(GG * NHEADS * 64 * SPGP, 256), 256, 0, stream>>>(qkvb, vtb);
  attn_kernel<<<CEILD(GG * NHEADS * QT, 8), 256, 0, stream>>>(qkvb, vtb, aob);
  // out-proj with residual: glob = h_nodes + (o @ out_w + out_b)
  gemm(aob, nullptr, nullptr, nullptr, nullptr, nullptr, 256, 1, out_wt,
       out_b, hn_f, glob, nullptr, nullptr, 0, NN, 256, 0);
  // fusion input = h_nodes + glob
  addcast_kernel<<<CEILD((long)NN * 256, 256), 256, 0, stream>>>(hn_f, glob, fusei, (long)NN * 256);
  gemm(fusei, nullptr, nullptr, nullptr, nullptr, nullptr, 256, 1, fu_w1t,
       fu_b1, nullptr, nullptr, tmpN, nullptr, 0, NN, 256, 1);
  gemm(tmpN, nullptr, nullptr, nullptr, nullptr, nullptr, 256, 1, fu_w2t,
       fu_b2, nullptr, nullptr, hfin, nullptr, 0, NN, 256, 0);

  // --- decoder: 3 type-MLPs, row-masked store into d_out ---
  for (int t = 0; t < 3; ++t) {
    gemm(hfin, nullptr, nullptr, nullptr, nullptr, nullptr, 256, 1, dec_w1t[t],
         dec_b1 + t * 256, nullptr, nullptr, tmpN, nullptr, 0, NN, 256, 1);
    gemm(tmpN, nullptr, nullptr, nullptr, nullptr, nullptr, 256, 1, dec_w2t[t],
         dec_b2 + t * 4, nullptr, out, nullptr, ntype, t, NN, OUTW, 0);
  }
  (void)in_sizes; (void)n_in; (void)out_size; (void)ws_size;
}
